// FeedbackTransformer_79061757985040
// MI455X (gfx1250) — matrix operand order repaired and checked by reading, not compiled, not run on hardware
//
#include <hip/hip_runtime.h>
#include <hip/hip_bf16.h>
#include <math.h>

#define S_ 128
#define B_ 32
#define D_ 512
#define H_ 8
#define DK_ 64
#define DFF_ 2048
#define L_ 4
#define P_ 4096
#define SCALE_ 0.125f  // 1/sqrt(64)

typedef __attribute__((ext_vector_type(16))) __bf16 vbf16;
typedef __attribute__((ext_vector_type(8)))  float  v8f;
typedef __attribute__((ext_vector_type(8)))  unsigned short u16x8;

union FragBF { vbf16 v; u16x8 h2[2]; unsigned short u[16]; };
union FragF  { v8f v; float f[8]; };

__device__ __forceinline__ unsigned short f2bf(float f) {
    unsigned int u = __float_as_uint(f);
    u += 0x7FFFu + ((u >> 16) & 1u);   // round-to-nearest-even
    return (unsigned short)(u >> 16);
}

__device__ __forceinline__ float wave_sum(float x) {
#pragma unroll
    for (int m = 16; m > 0; m >>= 1) x += __shfl_xor(x, m, 32);
    return x;
}
__device__ __forceinline__ float wave_max(float x) {
#pragma unroll
    for (int m = 16; m > 0; m >>= 1) x = fmaxf(x, __shfl_xor(x, m, 32));
    return x;
}

// ---------------------------------------------------------------------------
// WMMA 32x16 strip: C(32x16) += A(32 x 512 chunk in LDS, bf16) * W^T
// wt: bf16, row n holds K-major weights (ldw elements per row). kOff selects
// the K chunk matching the LDS tile. Accumulates 512/32 = 16 WMMA pairs.
// ---------------------------------------------------------------------------
__device__ __forceinline__ void strip_wmma(const unsigned short* ldsA,
                                           const unsigned short* wt, int ldw,
                                           int nBase, int kOff,
                                           v8f& c0, v8f& c1) {
    const int lane = threadIdx.x & 31;
    const int hl   = lane >> 4;     // wave half
    const int m    = lane & 15;
    const unsigned short* wrow = wt + (size_t)(nBase + m) * (size_t)ldw + kOff;
#pragma unroll 4
    for (int k0 = 0; k0 < 512; k0 += 32) {
        FragBF a0, a1, b;
        const unsigned short* ar0 = ldsA + m * 512 + k0 + hl * 8;
        const unsigned short* ar1 = ldsA + (16 + m) * 512 + k0 + hl * 8;
        a0.h2[0] = *(const u16x8*)(ar0);
        a0.h2[1] = *(const u16x8*)(ar0 + 16);
        a1.h2[0] = *(const u16x8*)(ar1);
        a1.h2[1] = *(const u16x8*)(ar1 + 16);
        const unsigned short* br = wrow + k0 + hl * 8;
        b.h2[0] = *(const u16x8*)(br);
        b.h2[1] = *(const u16x8*)(br + 16);
        c0 = __builtin_amdgcn_wmma_f32_16x16x32_bf16(false, a0.v, false, b.v,
                                                     (short)0, c0, false, false);
        c1 = __builtin_amdgcn_wmma_f32_16x16x32_bf16(false, a1.v, false, b.v,
                                                     (short)0, c1, false, false);
    }
}

// LN(32 x 512 f32) -> LDS bf16 (fused, 8 waves x 4 rows)
__device__ __forceinline__ void lds_fill_ln(unsigned short* ldsA, const float* A,
                                            const float* g, const float* beta) {
    const int wave = threadIdx.x >> 5;
    const int lane = threadIdx.x & 31;
    for (int r = wave; r < 32; r += 8) {
        const float* row = A + r * 512;
        float s = 0.f, ss = 0.f, vals[16];
#pragma unroll
        for (int i = 0; i < 16; i++) {
            float x = row[lane + i * 32];
            vals[i] = x; s += x; ss += x * x;
        }
        s = wave_sum(s); ss = wave_sum(ss);
        float mu = s * (1.0f / 512.0f);
        float var = ss * (1.0f / 512.0f) - mu * mu;
        float rstd = rsqrtf(var + 1e-5f);
#pragma unroll
        for (int i = 0; i < 16; i++) {
            int c = lane + i * 32;
            ldsA[r * 512 + c] = f2bf((vals[i] - mu) * rstd * g[c] + beta[c]);
        }
    }
    __syncthreads();
}

__device__ __forceinline__ void lds_fill_f32(unsigned short* ldsA, const float* A) {
    for (int i = threadIdx.x; i < 32 * 512; i += 256) ldsA[i] = f2bf(A[i]);
    __syncthreads();
}

__device__ __forceinline__ void lds_fill_bf16(unsigned short* ldsA,
                                              const unsigned short* A, int lda, int kOff) {
    for (int i = threadIdx.x; i < 32 * 512; i += 256) {
        int r = i >> 9, c = i & 511;
        ldsA[i] = A[(size_t)r * lda + kOff + c];
    }
    __syncthreads();
}

__device__ __forceinline__ float fuse_softmax(const float* fuse_w, int idx) {
    float fw[L_ + 1], mx = -1e30f;
#pragma unroll
    for (int i = 0; i < L_ + 1; i++) { fw[i] = fuse_w[i]; mx = fmaxf(mx, fw[i]); }
    float se = 0.f;
#pragma unroll
    for (int i = 0; i < L_ + 1; i++) { fw[i] = __expf(fw[i] - mx); se += fw[i]; }
    return fw[idx] / se;
}

// ---------------------------------------------------------------------------
// Kernels
// ---------------------------------------------------------------------------

// transpose+convert: in (L,K,N) f32 -> out (L,N,K) bf16
__global__ void __launch_bounds__(256) k_transcvt(const float* in, unsigned short* out,
                                                  int K, int N) {
    size_t total = (size_t)L_ * K * N;
    size_t i = (size_t)blockIdx.x * 256 + threadIdx.x;
    if (i >= total) return;
    size_t ln_sz = (size_t)K * N;
    int l = (int)(i / ln_sz);
    size_t r = i - (size_t)l * ln_sz;
    int n = (int)(r / K);
    int k = (int)(r - (size_t)n * K);
    out[i] = f2bf(in[((size_t)l * K + k) * N + n]);
}

// h = x_t ; acc = fuse[0]*x_t
__global__ void __launch_bounds__(256) k_init(const float* x_t, const float* fuse_w,
                                              float* h, float* acc) {
    int i = blockIdx.x * 256 + threadIdx.x;
    if (i >= B_ * D_) return;
    float f0 = fuse_softmax(fuse_w, 0);
    float x = x_t[i];
    h[i] = x; acc[i] = f0 * x;
}

// q = LN(h) @ wq  (grid 4, block 256)
__global__ void __launch_bounds__(256) k_ln_mm_q(const float* h, const float* g,
                                                 const float* b, const unsigned short* wt,
                                                 float* q) {
    __shared__ __align__(16) unsigned short ldsA[32 * 512];
    lds_fill_ln(ldsA, h, g, b);
    int wave = threadIdx.x >> 5;
    int nBase = blockIdx.x * 128 + wave * 16;
    v8f c0 = {}, c1 = {};
    strip_wmma(ldsA, wt, 512, nBase, 0, c0, c1);
    int lane = threadIdx.x & 31, hl = lane >> 4, m = lane & 15;
    FragF f0, f1; f0.v = c0; f1.v = c1;
#pragma unroll
    for (int r = 0; r < 8; r++) {
        q[(r + hl * 8) * 512 + nBase + m]      = f0.f[r];
        q[(16 + r + hl * 8) * 512 + nBase + m] = f1.f[r];
    }
}

// relative-position attention, one block per batch, one wave per head
__global__ void __launch_bounds__(256) k_attn(const float* q, const float* k_mem_l,
                                              const float* v_mem_l, const float* kpe_l,
                                              const float* kpb_l, const float* qpb_l,
                                              float* av, int t) {
    __shared__ float sc[H_][S_];
    int b = blockIdx.x;
    int hh = threadIdx.x >> 5;
    int lane = threadIdx.x & 31;
    int d0 = hh * DK_ + lane * 2;
    float q0 = q[b * D_ + d0], q1 = q[b * D_ + d0 + 1];
    float qa0 = q0 + qpb_l[hh * DK_ + lane * 2];
    float qa1 = q1 + qpb_l[hh * DK_ + lane * 2 + 1];
    for (int j = 0; j < S_; j++) {
        const float* kr = k_mem_l + ((size_t)j * B_ + b) * D_ + d0;
        float part = qa0 * kr[0] + qa1 * kr[1];
        int didx = t - j - 1;
        didx = didx < 0 ? 0 : (didx > S_ - 1 ? S_ - 1 : didx);
        int pidx = P_ - 1 - didx;
        const float* pe = kpe_l + ((size_t)pidx * H_ + hh) * DK_ + lane * 2;
        part += q0 * pe[0] + q1 * pe[1];
        part = wave_sum(part);
        if (lane == 0)
            sc[hh][j] = (j < t) ? (part + kpb_l[pidx * H_ + hh]) * SCALE_ : -1e9f;
    }
    __syncthreads();
    float mx = -1e30f;
#pragma unroll
    for (int i = 0; i < 4; i++) mx = fmaxf(mx, sc[hh][lane + i * 32]);
    mx = wave_max(mx);
    float se = 0.f;
#pragma unroll
    for (int i = 0; i < 4; i++) se += __expf(sc[hh][lane + i * 32] - mx);
    se = wave_sum(se);
    float inv = 1.0f / se;
    float a0 = 0.f, a1 = 0.f;
    for (int j = 0; j < S_; j++) {
        float w = __expf(sc[hh][j] - mx) * inv;
        const float* vr = v_mem_l + ((size_t)j * B_ + b) * D_ + d0;
        a0 += w * vr[0]; a1 += w * vr[1];
    }
    av[b * D_ + d0] = a0;
    av[b * D_ + d0 + 1] = a1;
}

// h += gate * (av @ wo + bo)   (grid 4)
__global__ void __launch_bounds__(256) k_mm_wo(const float* av, const unsigned short* wt,
                                               const float* bias, float* h, int gate) {
    __shared__ __align__(16) unsigned short ldsA[32 * 512];
    lds_fill_f32(ldsA, av);
    int wave = threadIdx.x >> 5;
    int nBase = blockIdx.x * 128 + wave * 16;
    v8f c0 = {}, c1 = {};
    strip_wmma(ldsA, wt, 512, nBase, 0, c0, c1);
    if (!gate) return;
    int lane = threadIdx.x & 31, hl = lane >> 4, m = lane & 15;
    int col = nBase + m;
    float bb = bias[col];
    FragF f0, f1; f0.v = c0; f1.v = c1;
#pragma unroll
    for (int r = 0; r < 8; r++) {
        h[(r + hl * 8) * 512 + col]      += f0.f[r] + bb;
        h[(16 + r + hl * 8) * 512 + col] += f1.f[r] + bb;
    }
}

// ff1 = relu(LN(h) @ w1 + b1) stored bf16 (grid 16)
__global__ void __launch_bounds__(256) k_ln_mm_ff1(const float* h, const float* g,
                                                   const float* b, const unsigned short* wt,
                                                   const float* bias, unsigned short* out) {
    __shared__ __align__(16) unsigned short ldsA[32 * 512];
    lds_fill_ln(ldsA, h, g, b);
    int wave = threadIdx.x >> 5;
    int nBase = blockIdx.x * 128 + wave * 16;
    v8f c0 = {}, c1 = {};
    strip_wmma(ldsA, wt, 512, nBase, 0, c0, c1);
    int lane = threadIdx.x & 31, hl = lane >> 4, m = lane & 15;
    int col = nBase + m;
    float bb = bias[col];
    FragF f0, f1; f0.v = c0; f1.v = c1;
#pragma unroll
    for (int r = 0; r < 8; r++) {
        out[(r + hl * 8) * DFF_ + col]      = f2bf(fmaxf(f0.f[r] + bb, 0.f));
        out[(16 + r + hl * 8) * DFF_ + col] = f2bf(fmaxf(f1.f[r] + bb, 0.f));
    }
}

// h += ff1 @ w2 + b2 ; acc += fuse[l+1]*h ; optionally res_t = h  (grid 4)
__global__ void __launch_bounds__(256) k_mm_ff2(const unsigned short* ff1,
                                                const unsigned short* wt, const float* bias,
                                                float* h, float* acc, const float* fuse_w,
                                                int lplus1, float* res_t) {
    __shared__ __align__(16) unsigned short ldsA[32 * 512];
    int wave = threadIdx.x >> 5;
    int nBase = blockIdx.x * 128 + wave * 16;
    v8f c0 = {}, c1 = {};
    for (int kc = 0; kc < 4; kc++) {
        lds_fill_bf16(ldsA, ff1, DFF_, kc * 512);
        strip_wmma(ldsA, wt, DFF_, nBase, kc * 512, c0, c1);
        __syncthreads();   // all waves done reading before next chunk overwrites
    }
    float fuse = fuse_softmax(fuse_w, lplus1);
    int lane = threadIdx.x & 31, hl = lane >> 4, m = lane & 15;
    int col = nBase + m;
    float bb = bias[col];
    FragF f0, f1; f0.v = c0; f1.v = c1;
#pragma unroll
    for (int r = 0; r < 8; r++) {
        int i0 = (r + hl * 8) * 512 + col;
        int i1 = (16 + r + hl * 8) * 512 + col;
        float h0 = h[i0] + f0.f[r] + bb;
        float h1 = h[i1] + f1.f[r] + bb;
        h[i0] = h0; h[i1] = h1;
        acc[i0] += fuse * h0; acc[i1] += fuse * h1;
        if (res_t) { res_t[i0] = h0; res_t[i1] = h1; }
    }
}

// k_mem[l,t] / v_mem[l,t] = acc @ wk[l] / acc @ wv[l] + bv[l]  (grid 32)
__global__ void __launch_bounds__(256) k_mm_kv(const float* acc, const unsigned short* wkt,
                                               const unsigned short* wvt, const float* bv,
                                               float* k_mem, float* v_mem, int t) {
    __shared__ __align__(16) unsigned short ldsA[32 * 512];
    lds_fill_f32(ldsA, acc);
    int bx = blockIdx.x;
    int chunk = bx & 3, l = (bx >> 2) & 3, which = bx >> 4;
    const unsigned short* wt = (which ? wvt : wkt) + (size_t)l * 512 * 512;
    int wave = threadIdx.x >> 5;
    int nBase = chunk * 128 + wave * 16;
    v8f c0 = {}, c1 = {};
    strip_wmma(ldsA, wt, 512, nBase, 0, c0, c1);
    int lane = threadIdx.x & 31, hl = lane >> 4, m = lane & 15;
    int col = nBase + m;
    float bb = which ? bv[l * 512 + col] : 0.f;
    float* dst = (which ? v_mem : k_mem) + ((size_t)(l * S_ + t) * B_) * D_;
    FragF f0, f1; f0.v = c0; f1.v = c1;
#pragma unroll
    for (int r = 0; r < 8; r++) {
        dst[(r + hl * 8) * 512 + col]      = f0.f[r] + bb;
        dst[(16 + r + hl * 8) * 512 + col] = f1.f[r] + bb;
    }
}

// final LN over res (S*B rows x 512)
__global__ void __launch_bounds__(256) k_final_ln(const float* res, const float* g,
                                                  const float* b, float* out) {
    int row = blockIdx.x * 8 + (threadIdx.x >> 5);
    int lane = threadIdx.x & 31;
    const float* rr = res + (size_t)row * 512;
    float s = 0.f, ss = 0.f, vals[16];
#pragma unroll
    for (int i = 0; i < 16; i++) {
        float x = rr[lane + i * 32];
        vals[i] = x; s += x; ss += x * x;
    }
    s = wave_sum(s); ss = wave_sum(ss);
    float mu = s * (1.0f / 512.0f);
    float rstd = rsqrtf(ss * (1.0f / 512.0f) - mu * mu + 1e-5f);
#pragma unroll
    for (int i = 0; i < 16; i++) {
        int c = lane + i * 32;
        out[(size_t)row * 512 + c] = (vals[i] - mu) * rstd * g[c] + b[c];
    }
}

// ---------------------------------------------------------------------------
extern "C" void kernel_launch(void* const* d_in, const int* in_sizes, int n_in,
                              void* d_out, int out_size, void* d_ws, size_t ws_size,
                              hipStream_t stream) {
    const float* x_seq   = (const float*)d_in[0];
    const float* ln_sa_g = (const float*)d_in[1];
    const float* ln_sa_b = (const float*)d_in[2];
    const float* wq      = (const float*)d_in[3];
    const float* wk      = (const float*)d_in[4];
    const float* wv      = (const float*)d_in[5];
    const float* bv      = (const float*)d_in[6];
    const float* kpe     = (const float*)d_in[7];
    const float* kpb     = (const float*)d_in[8];
    const float* qpb     = (const float*)d_in[9];
    const float* wo      = (const float*)d_in[10];
    const float* bo      = (const float*)d_in[11];
    const float* ln_ff_g = (const float*)d_in[12];
    const float* ln_ff_b = (const float*)d_in[13];
    const float* w1      = (const float*)d_in[14];
    const float* b1      = (const float*)d_in[15];
    const float* w2      = (const float*)d_in[16];
    const float* b2      = (const float*)d_in[17];
    const float* fuse_w  = (const float*)d_in[18];
    const float* ln_o_g  = (const float*)d_in[19];
    const float* ln_o_b  = (const float*)d_in[20];

    char* p = (char*)d_ws;
    auto carve = [&](size_t bytes) -> char* {
        char* r = p; p += (bytes + 255) & ~(size_t)255; return r;
    };
    unsigned short* wq_t = (unsigned short*)carve((size_t)L_ * D_ * D_ * 2);
    unsigned short* wk_t = (unsigned short*)carve((size_t)L_ * D_ * D_ * 2);
    unsigned short* wv_t = (unsigned short*)carve((size_t)L_ * D_ * D_ * 2);
    unsigned short* wo_t = (unsigned short*)carve((size_t)L_ * D_ * D_ * 2);
    unsigned short* w1_t = (unsigned short*)carve((size_t)L_ * D_ * DFF_ * 2);
    unsigned short* w2_t = (unsigned short*)carve((size_t)L_ * DFF_ * D_ * 2);
    float* k_mem = (float*)carve((size_t)L_ * S_ * B_ * D_ * 4);
    float* v_mem = (float*)carve((size_t)L_ * S_ * B_ * D_ * 4);
    float* res   = (float*)carve((size_t)S_ * B_ * D_ * 4);
    float* hbuf  = (float*)carve((size_t)B_ * D_ * 4);
    float* accb  = (float*)carve((size_t)B_ * D_ * 4);
    float* qbuf  = (float*)carve((size_t)B_ * D_ * 4);
    float* avbuf = (float*)carve((size_t)B_ * D_ * 4);
    unsigned short* ff1b = (unsigned short*)carve((size_t)B_ * DFF_ * 2);

    // weight prep: transpose + bf16
    size_t tdd = (size_t)L_ * D_ * D_;
    k_transcvt<<<(int)((tdd + 255) / 256), 256, 0, stream>>>(wq, wq_t, D_, D_);
    k_transcvt<<<(int)((tdd + 255) / 256), 256, 0, stream>>>(wk, wk_t, D_, D_);
    k_transcvt<<<(int)((tdd + 255) / 256), 256, 0, stream>>>(wv, wv_t, D_, D_);
    k_transcvt<<<(int)((tdd + 255) / 256), 256, 0, stream>>>(wo, wo_t, D_, D_);
    size_t tdf = (size_t)L_ * D_ * DFF_;
    k_transcvt<<<(int)((tdf + 255) / 256), 256, 0, stream>>>(w1, w1_t, D_, DFF_);
    k_transcvt<<<(int)((tdf + 255) / 256), 256, 0, stream>>>(w2, w2_t, DFF_, D_);

    for (int t = 0; t < S_; t++) {
        k_init<<<64, 256, 0, stream>>>(x_seq + (size_t)t * B_ * D_, fuse_w, hbuf, accb);
        for (int l = 0; l < L_; l++) {
            k_ln_mm_q<<<4, 256, 0, stream>>>(hbuf, ln_sa_g + l * D_, ln_sa_b + l * D_,
                                             wq_t + (size_t)l * D_ * D_, qbuf);
            k_attn<<<32, 256, 0, stream>>>(qbuf,
                                           k_mem + (size_t)l * S_ * B_ * D_,
                                           v_mem + (size_t)l * S_ * B_ * D_,
                                           kpe + (size_t)l * P_ * H_ * DK_,
                                           kpb + (size_t)l * P_ * H_,
                                           qpb + (size_t)l * H_ * DK_, avbuf, t);
            k_mm_wo<<<4, 256, 0, stream>>>(avbuf, wo_t + (size_t)l * D_ * D_,
                                           bo + l * D_, hbuf, t > 0 ? 1 : 0);
            k_ln_mm_ff1<<<16, 256, 0, stream>>>(hbuf, ln_ff_g + l * D_, ln_ff_b + l * D_,
                                                w1_t + (size_t)l * DFF_ * D_,
                                                b1 + l * DFF_, ff1b);
            k_mm_ff2<<<4, 256, 0, stream>>>(ff1b, w2_t + (size_t)l * D_ * DFF_,
                                            b2 + l * D_, hbuf, accb, fuse_w, l + 1,
                                            (l == L_ - 1) ? (res + (size_t)t * B_ * D_)
                                                          : nullptr);
        }
        k_mm_kv<<<32, 256, 0, stream>>>(accb, wk_t, wv_t, bv, k_mem, v_mem, t);
    }
    k_final_ln<<<S_ * B_ / 8, 256, 0, stream>>>(res, ln_o_g, ln_o_b, (float*)d_out);
}
